// TADRModel_78606491451450
// MI455X (gfx1250) — matrix operand order repaired and checked by reading, not compiled, not run on hardware
//
#include <hip/hip_runtime.h>
#include <hip/hip_bf16.h>

// ---------------- problem constants (match reference) ----------------
#define BB 8
#define SS 2048
#define HH 1024
#define DD 256
#define KK 10
#define TT 128
#define RR 64
#define EPS 1e-5f

#define MT 32                 // tokens per block tile
#define NTHREADS 256          // 8 waves of 32

// LDS carve-up (bytes)
#define SM_XBF   0                                  // [MT][HH] bf16  = 65536
#define SM_DNBF  (SM_XBF + MT*HH*2)                 // [MT][DD] bf16  = 16384
#define SM_OUT   (SM_DNBF + MT*DD*2)                // [MT][HH] f32   = 131072
#define SM_RSUM  (SM_OUT + MT*HH*4)                 // [MT] f32
#define SM_RSSQ  (SM_RSUM + MT*4)                   // [MT] f32
#define SMEM_BYTES (SM_RSSQ + MT*4)                 // 213248 total

// workspace layout (bytes)
#define WS_RW    0                                  // [B][K] f32 routing weights
#define WS_WDSW  4096                               // swizzled Wd bf16: 5 MB
#define WDSW_DWORDS (KK * 16 * 32 * 32 * 8)         // 1,310,720 u32
#define WS_WUSW  (WS_WDSW + WDSW_DWORDS * 4)        // swizzled Wu bf16: 5 MB
#define WUSW_DWORDS (KK * 64 * 8 * 32 * 8)          // 1,310,720 u32

typedef __attribute__((ext_vector_type(16))) __bf16 v16bf;
typedef __attribute__((ext_vector_type(8)))  float  v8f;

union Frag { v16bf v; unsigned u[8]; uint4 q[2]; };

__device__ __forceinline__ unsigned pack2bf(float lo, float hi) {
  unsigned a = __float_as_uint(lo);
  unsigned b = __float_as_uint(hi);
  a += 0x7FFFu + ((a >> 16) & 1u);      // round-to-nearest-even f32 -> bf16
  b += 0x7FFFu + ((b >> 16) & 1u);
  return (a >> 16) | (b & 0xFFFF0000u);
}

__device__ __forceinline__ unsigned short f2bf(float f) {
  unsigned a = __float_as_uint(f);
  a += 0x7FFFu + ((a >> 16) & 1u);
  return (unsigned short)(a >> 16);
}

// ---------------- router: routing_weights[B][K] = softmax(MLP(emb)) --------
__global__ void __launch_bounds__(256)
router_kernel(const float* __restrict__ emb, const float* __restrict__ Wr1,
              const float* __restrict__ br1, const float* __restrict__ Wr2,
              const float* __restrict__ br2, float* __restrict__ w_out) {
  __shared__ float h[BB][RR];
  __shared__ float logits[BB][KK];
  const int t = threadIdx.x;
  const int b = t >> 5;
  const int lane = t & 31;
  #pragma unroll
  for (int rr = 0; rr < 2; ++rr) {
    const int r = lane * 2 + rr;
    float acc = br1[r];
    for (int k = 0; k < TT; ++k) acc += emb[b * TT + k] * Wr1[k * RR + r];
    h[b][r] = fmaxf(acc, 0.0f);
  }
  __syncthreads();
  if (lane < KK) {
    float acc = br2[lane];
    #pragma unroll 8
    for (int r = 0; r < RR; ++r) acc += h[b][r] * Wr2[r * KK + lane];
    logits[b][lane] = acc;
  }
  __syncthreads();
  if (lane < KK) {
    float m = -1e30f;
    for (int k = 0; k < KK; ++k) m = fmaxf(m, logits[b][k]);
    float den = 0.0f;
    for (int k = 0; k < KK; ++k) den += __expf(logits[b][k] - m);
    w_out[b * KK + lane] = __expf(logits[b][lane] - m) / den;
  }
}

// ---------------- weight pre-swizzle: f32 [K][kk][n] -> bf16 B-fragments ----
// Fragment layout: [k][ntg][kstep][lane][8 dwords]; lane holds N = ntg*16 +
// (lane&15), K = kstep*32 + (lane<16?0:8) + (v<4 ? 2v : 16+2(v-4)) .. +1  (ISA 7.12.2 B layout).
__global__ void __launch_bounds__(256)
prep_weights_kernel(const float* __restrict__ W, unsigned* __restrict__ dst,
                    int ntg_count, int nst_count, int nd) {
  const long idx = (long)blockIdx.x * 256 + threadIdx.x;
  const int v    = (int)(idx & 7);
  const int lane = (int)((idx >> 3) & 31);
  long fs = idx >> 8;
  const int step = (int)(fs % nst_count); fs /= nst_count;
  const int ntg  = (int)(fs % ntg_count); fs /= ntg_count;
  const int k    = (int)fs;
  if (k >= KK) return;
  const int n  = ntg * 16 + (lane & 15);
  const int kk = step * 32 + ((lane < 16) ? 0 : 8) + ((v < 4) ? 2 * v : 16 + 2 * (v - 4));
  const size_t base = (size_t)k * (size_t)nst_count * 32u * (size_t)nd;
  dst[idx] = pack2bf(W[base + (size_t)kk * nd + n],
                     W[base + (size_t)(kk + 1) * nd + n]);
}

// ---------------- main fused adapter kernel --------------------------------
// out[b,s,:] = x + sum_k w[b,k] * LN_k( relu(x Wd_k + bd_k) Wu_k + bu_k )
__global__ void __launch_bounds__(NTHREADS)
adapter_kernel(const float* __restrict__ x,
               const unsigned* __restrict__ wdsw, const float* __restrict__ bd,
               const unsigned* __restrict__ wusw, const float* __restrict__ bu,
               const float* __restrict__ gamma, const float* __restrict__ beta,
               const float* __restrict__ rw, float* __restrict__ out) {
  extern __shared__ char smem[];
  unsigned short* xbf  = (unsigned short*)(smem + SM_XBF);   // [MT][HH] bf16
  unsigned short* dnbf = (unsigned short*)(smem + SM_DNBF);  // [MT][DD] bf16
  float*          oacc = (float*)(smem + SM_OUT);            // [MT][HH] f32
  float*          rsum = (float*)(smem + SM_RSUM);           // [MT] (sum -> mu)
  float*          rssq = (float*)(smem + SM_RSSQ);           // [MT] (sumsq -> rs)

  const int tid  = threadIdx.x;
  const int wave = tid >> 5;
  const int lane = tid & 31;
  const long tok0 = (long)blockIdx.x * MT;
  const int  b    = (int)(tok0 >> 11);         // batch (uniform: 32 | 2048)

  // ---- stage x tile f32 -> LDS via async copy (reuses oacc region) ----
  {
    const float* src = x + tok0 * HH;
    for (int c = tid; c < (MT * HH) / 4; c += NTHREADS) {   // 32 iters, full EXEC
      const unsigned lds_off = (unsigned)(SM_OUT + 16 * c);
      const float* g = src + 4 * c;
      asm volatile("global_load_async_to_lds_b128 %0, %1, off"
                   :: "v"(lds_off), "v"(g) : "memory");
    }
    asm volatile("s_wait_asynccnt 0" ::: "memory");
    __syncthreads();
    const float* xf32 = oacc;
    for (int i = tid; i < MT * HH; i += NTHREADS) xbf[i] = f2bf(xf32[i]);
    __syncthreads();
    for (int i = tid; i < MT * HH; i += NTHREADS) oacc[i] = 0.0f;
  }
  __syncthreads();

  // WMMA lane-layout constants (ISA 7.12.2, 16-bit operands, wave32)
  const int mrow   = lane & 15;
  const int akbase = (lane < 16) ? 0 : 8;    // A: interleaved K halves
  const int crow0  = (lane < 16) ? 0 : 8;    // C/D: rows v / v+8

  // loop-invariant per-lane base pointers (per-step/tile offsets fold into
  // the 24-bit global / 16-bit DS instruction offsets)
  const char* paX = (const char*)(xbf + mrow * HH + akbase);   // + mt*32768 + step*64
  const char* paD = (const char*)(dnbf + mrow * DD + akbase);  // + mt*8192  + step*64

  for (int k = 0; k < KK; ++k) {
    const float wk = rw[b * KK + k];

    // ====== down GEMM: dn[32x256] = xbf[32x1024] * Wd_k[1024x256] ======
    v8f dacc[2][2];
    #pragma unroll
    for (int mt = 0; mt < 2; ++mt)
      #pragma unroll
      for (int nt = 0; nt < 2; ++nt)
        #pragma unroll
        for (int e = 0; e < 8; ++e) dacc[mt][nt][e] = 0.0f;

    // frag byte address: ((k*16+ntg)*32+step)*1024 + lane*32
    const char* pbD = (const char*)wdsw
                    + ((size_t)(k * 16 + wave * 2) * 32) * 1024
                    + (size_t)lane * 32;

    #pragma unroll 2
    for (int step = 0; step < 32; ++step) {
      Frag afr[2];
      #pragma unroll
      for (int mt = 0; mt < 2; ++mt) {   // A fragment = 2x ds_load_b128
        const char* pa = paX + mt * (16 * HH * 2) + step * 64;
        afr[mt].q[0] = *(const uint4*)(pa);
        afr[mt].q[1] = *(const uint4*)(pa + 32);   // K = kbase+16..kbase+23
      }
      #pragma unroll
      for (int nt = 0; nt < 2; ++nt) {   // B fragment = 2x global_load_b128
        const char* pb = pbD + nt * (32 * 1024) + step * 1024;
        Frag bfr;
        bfr.q[0] = *(const uint4*)(pb);
        bfr.q[1] = *(const uint4*)(pb + 16);
        #pragma unroll
        for (int mt = 0; mt < 2; ++mt)
          dacc[mt][nt] = __builtin_amdgcn_wmma_f32_16x16x32_bf16(
              false, afr[mt].v, false, bfr.v, (short)0, dacc[mt][nt],
              false, false);
      }
      __builtin_prefetch(pbD + step * 1024 + 2048, 0, 1);
    }

    // relu(dn + bd) -> LDS bf16 (A operand of up GEMM)
    #pragma unroll
    for (int mt = 0; mt < 2; ++mt)
      #pragma unroll
      for (int nt = 0; nt < 2; ++nt) {
        const int col = wave * 32 + nt * 16 + (lane & 15);
        const float bias = bd[k * DD + col];
        #pragma unroll
        for (int v = 0; v < 8; ++v) {
          const int row = mt * 16 + crow0 + v;
          dnbf[row * DD + col] = f2bf(fmaxf(dacc[mt][nt][v] + bias, 0.0f));
        }
      }
    if (tid < MT) { rsum[tid] = 0.0f; rssq[tid] = 0.0f; }
    __syncthreads();

    // ====== up GEMM: up[32x1024] = dn[32x256] * Wu_k[256x1024] ======
    v8f uacc[2][8];
    #pragma unroll
    for (int mt = 0; mt < 2; ++mt)
      #pragma unroll
      for (int nt = 0; nt < 8; ++nt)
        #pragma unroll
        for (int e = 0; e < 8; ++e) uacc[mt][nt][e] = 0.0f;

    // frag byte address: ((k*64+ntg)*8+step)*1024 + lane*32
    const char* pbU = (const char*)wusw
                    + ((size_t)(k * 64 + wave * 8) * 8) * 1024
                    + (size_t)lane * 32;

    #pragma unroll 2
    for (int step = 0; step < 8; ++step) {
      Frag afr[2];
      #pragma unroll
      for (int mt = 0; mt < 2; ++mt) {
        const char* pa = paD + mt * (16 * DD * 2) + step * 64;
        afr[mt].q[0] = *(const uint4*)(pa);
        afr[mt].q[1] = *(const uint4*)(pa + 32);
      }
      #pragma unroll
      for (int nt = 0; nt < 8; ++nt) {
        const char* pb = pbU + nt * (8 * 1024) + step * 1024;
        Frag bfr;
        bfr.q[0] = *(const uint4*)(pb);
        bfr.q[1] = *(const uint4*)(pb + 16);
        #pragma unroll
        for (int mt = 0; mt < 2; ++mt)
          uacc[mt][nt] = __builtin_amdgcn_wmma_f32_16x16x32_bf16(
              false, afr[mt].v, false, bfr.v, (short)0, uacc[mt][nt],
              false, false);
      }
      __builtin_prefetch(pbU + step * 1024 + 1024, 0, 1);
    }

    // bias + per-row partial sums: sum over the 8 N-tiles in registers first,
    // then one 16-lane shuffle reduction per (mt, v) row.
    float bbias[8];
    #pragma unroll
    for (int nt = 0; nt < 8; ++nt)
      bbias[nt] = bu[k * HH + wave * 128 + nt * 16 + (lane & 15)];
    #pragma unroll
    for (int mt = 0; mt < 2; ++mt)
      #pragma unroll
      for (int v = 0; v < 8; ++v) {
        float s = 0.0f, s2 = 0.0f;
        #pragma unroll
        for (int nt = 0; nt < 8; ++nt) {
          const float val = uacc[mt][nt][v] + bbias[nt];
          uacc[mt][nt][v] = val;
          s += val;
          s2 += val * val;
        }
        #pragma unroll
        for (int off = 1; off < 16; off <<= 1) {
          s  += __shfl_xor(s,  off, 32);
          s2 += __shfl_xor(s2, off, 32);
        }
        if ((lane & 15) == 0) {
          const int row = mt * 16 + crow0 + v;
          atomicAdd(&rsum[row], s);
          atomicAdd(&rssq[row], s2);
        }
      }
    __syncthreads();

    // finalize per-row statistics ONCE (rsum -> mu, rssq -> rsqrt(var+eps)):
    // removes 3 VALU + 1 transcendental per element from the accumulate loop
    if (tid < MT) {
      const float invH = 1.0f / (float)HH;
      const float mu  = rsum[tid] * invH;
      const float var = rssq[tid] * invH - mu * mu;
      rsum[tid] = mu;
      rssq[tid] = __frsqrt_rn(var + EPS);
    }
    __syncthreads();

    // LayerNorm + routed accumulate (single owner thread per element);
    // wk folded into gamma/beta: wk*ln = (val-mu)*rs*(g*wk) + be*wk
    #pragma unroll
    for (int mt = 0; mt < 2; ++mt)
      #pragma unroll
      for (int nt = 0; nt < 8; ++nt) {
        const int col = wave * 128 + nt * 16 + (lane & 15);
        const float gw = gamma[k * HH + col] * wk;
        const float bw = beta[k * HH + col] * wk;
        #pragma unroll
        for (int v = 0; v < 8; ++v) {
          const int row = mt * 16 + crow0 + v;
          const float t = (uacc[mt][nt][v] - rsum[row]) * rssq[row];
          oacc[row * HH + col] += t * gw + bw;
        }
      }
    __syncthreads();   // protect dnbf/rsum before next adapter
  }

  // ---- residual + store: out = x + sum_k w_k * LN_k (softmax sums to 1) ----
  for (int i = tid; i < MT * HH; i += NTHREADS) {
    out[tok0 * HH + i] = x[tok0 * HH + i] + oacc[i];
  }
}

// ---------------- launch ---------------------------------------------------
extern "C" void kernel_launch(void* const* d_in, const int* in_sizes, int n_in,
                              void* d_out, int out_size, void* d_ws, size_t ws_size,
                              hipStream_t stream) {
  const float* x     = (const float*)d_in[0];
  const float* emb   = (const float*)d_in[1];
  const float* Wd    = (const float*)d_in[2];
  const float* bd    = (const float*)d_in[3];
  const float* Wu    = (const float*)d_in[4];
  const float* bu    = (const float*)d_in[5];
  const float* gamma = (const float*)d_in[6];
  const float* beta  = (const float*)d_in[7];
  const float* Wr1   = (const float*)d_in[8];
  const float* br1   = (const float*)d_in[9];
  const float* Wr2   = (const float*)d_in[10];
  const float* br2   = (const float*)d_in[11];
  float* out = (float*)d_out;

  float*    rw   = (float*)((char*)d_ws + WS_RW);
  unsigned* wdsw = (unsigned*)((char*)d_ws + WS_WDSW);
  unsigned* wusw = (unsigned*)((char*)d_ws + WS_WUSW);

  router_kernel<<<1, 256, 0, stream>>>(emb, Wr1, br1, Wr2, br2, rw);

  // pre-swizzle weights into bf16 WMMA B-fragment layout (L2-resident, 10 MB)
  prep_weights_kernel<<<WDSW_DWORDS / 256, 256, 0, stream>>>(Wd, wdsw, 16, 32, DD);
  prep_weights_kernel<<<WUSW_DWORDS / 256, 256, 0, stream>>>(Wu, wusw, 64, 8, HH);

  hipFuncSetAttribute(reinterpret_cast<const void*>(adapter_kernel),
                      hipFuncAttributeMaxDynamicSharedMemorySize, SMEM_BYTES);

  const int nblocks = (BB * SS) / MT;   // 512 token tiles
  adapter_kernel<<<nblocks, NTHREADS, SMEM_BYTES, stream>>>(
      x, wdsw, bd, wusw, bu, gamma, beta, rw, out);
}
